// TransformerBlock_37864431681712
// MI455X (gfx1250) — matrix operand order repaired and checked by reading, not compiled, not run on hardware
//
#include <hip/hip_runtime.h>

// ---------------------------------------------------------------------------
// Transformer block forward for MI455X (gfx1250), bf16 WMMA mixed precision.
// ---------------------------------------------------------------------------

#define BATCH 2
#define SEQL  2048
#define DMODEL 2048
#define NHEAD 16
#define HDIM  128
#define DFF   8192
#define MROWS (BATCH * SEQL)   // 4096

typedef __attribute__((ext_vector_type(16))) __bf16 v16bf;
typedef __attribute__((ext_vector_type(8)))  __bf16 v8bf;
typedef __attribute__((ext_vector_type(8)))  float  v8f;

union BF16x16 { v16bf v; v8bf h[2]; };

__device__ __forceinline__ __bf16 f2bf(float f) {
    union { float f; unsigned int u; } a; a.f = f;
    unsigned int r = a.u + 0x7FFFu + ((a.u >> 16) & 1u);   // round-nearest-even
    union { unsigned short s; __bf16 b; } o; o.s = (unsigned short)(r >> 16);
    return o.b;
}

__device__ __forceinline__ v16bf load16(const __bf16* p0, const __bf16* p1) {
    BF16x16 r;
    r.h[0] = *(const v8bf*)p0;
    r.h[1] = *(const v8bf*)p1;
    return r.v;
}

__device__ __forceinline__ float gelu_exact(float x) {
    return 0.5f * x * (1.0f + erff(x * 0.70710678118654752440f));
}

// ---------------------------------------------------------------------------
// fp32 -> bf16 convert (weights)
// ---------------------------------------------------------------------------
__global__ __launch_bounds__(256) void cvt_f32_bf16(const float* __restrict__ in,
                                                    __bf16* __restrict__ out, size_t n) {
    size_t i = (size_t)blockIdx.x * 256 + threadIdx.x;
    if (i < n) out[i] = f2bf(in[i]);
}

// ---------------------------------------------------------------------------
// LayerNorm: fp32 row -> bf16 row. One block per row, 256 threads.
// ---------------------------------------------------------------------------
__global__ __launch_bounds__(256) void layernorm_bf16(const float* __restrict__ x,
                                                      const float* __restrict__ w,
                                                      const float* __restrict__ b,
                                                      __bf16* __restrict__ out, int C) {
    __shared__ float sh[2][8];
    const int lane = threadIdx.x & 31;
    const int wave = threadIdx.x >> 5;
    const float* xr = x + (size_t)blockIdx.x * C;

    float s = 0.f, s2 = 0.f;
    for (int i = threadIdx.x; i < C; i += 256) {
        float v = xr[i]; s += v; s2 += v * v;
    }
    #pragma unroll
    for (int o = 16; o > 0; o >>= 1) { s += __shfl_xor(s, o, 32); s2 += __shfl_xor(s2, o, 32); }
    if (lane == 0) { sh[0][wave] = s; sh[1][wave] = s2; }
    __syncthreads();
    float ts = 0.f, ts2 = 0.f;
    #pragma unroll
    for (int wv = 0; wv < 8; wv++) { ts += sh[0][wv]; ts2 += sh[1][wv]; }
    const float mu = ts / (float)C;
    const float var = ts2 / (float)C - mu * mu;
    const float rstd = rsqrtf(var + 1e-5f);

    __bf16* orow = out + (size_t)blockIdx.x * C;
    for (int i = threadIdx.x; i < C; i += 256)
        orow[i] = f2bf((xr[i] - mu) * rstd * w[i] + b[i]);
}

// ---------------------------------------------------------------------------
// WMMA GEMM: out = act(A[M,K] @ W[N,K]^T + bias) (+ resid). A,W bf16, acc f32.
// Block = 8 waves (2 M x 4 N), block tile 128x128, wave tile 64x32 (4x2 WMMAs).
// Requires M%128==0, N%128==0, K%32==0 (all shapes here comply).
// ---------------------------------------------------------------------------
template <int ACT, bool RES, bool OUTF32, bool OUTBF16>
__global__ __launch_bounds__(256, 1) void gemm_wmma_bf16(
    const __bf16* __restrict__ A, const __bf16* __restrict__ W,
    const float* __restrict__ bias, const float* __restrict__ resid,
    float* __restrict__ outF, __bf16* __restrict__ outB,
    int M, int N, int K) {
    const int lane = threadIdx.x & 31;
    const int wave = threadIdx.x >> 5;
    const int l16 = lane & 15, lh = lane >> 4;
    const int bm = blockIdx.y * 128 + (wave & 1) * 64;   // wave M base
    const int bn = blockIdx.x * 128 + (wave >> 1) * 32;  // wave N base

    v8f acc[4][2] = {};

    // A-fragment: lane holds row (bm+i*16+l16); K halves at lh*8 and 16+lh*8.
    const __bf16* aBase = A + (size_t)(bm + l16) * K + lh * 8;
    // B-fragment: lane holds col (bn+j*16+l16); K halves at lh*8 and 16+lh*8.
    const __bf16* wBase = W + (size_t)(bn + l16) * K + lh * 8;

    for (int k0 = 0; k0 < K; k0 += 32) {
        if (k0 + 128 < K) {
            __builtin_prefetch(aBase + k0 + 128, 0, 1);
            __builtin_prefetch(wBase + k0 + 128, 0, 1);
        }
        v16bf af[4], wf[2];
        #pragma unroll
        for (int i = 0; i < 4; i++) {
            const __bf16* p = aBase + (size_t)i * 16 * K + k0;
            af[i] = load16(p, p + 16);
        }
        #pragma unroll
        for (int j = 0; j < 2; j++) {
            const __bf16* p = wBase + (size_t)j * 16 * K + k0;
            wf[j] = load16(p, p + 16);
        }
        #pragma unroll
        for (int i = 0; i < 4; i++)
            #pragma unroll
            for (int j = 0; j < 2; j++)
                acc[i][j] = __builtin_amdgcn_wmma_f32_16x16x32_bf16(
                    false, af[i], false, wf[j], (short)0, acc[i][j], false, false);
    }

    // Epilogue. C-layout: lane -> col l16; VGPR v -> row v + 8*lh.
    #pragma unroll
    for (int i = 0; i < 4; i++) {
        #pragma unroll
        for (int j = 0; j < 2; j++) {
            #pragma unroll
            for (int v = 0; v < 8; v++) {
                const int row = bm + i * 16 + v + lh * 8;
                const int col = bn + j * 16 + l16;
                float val = acc[i][j][v] + bias[col];
                if (ACT == 1) val = gelu_exact(val);
                if (RES) val += resid[(size_t)row * N + col];
                if (OUTF32) outF[(size_t)row * N + col] = val;
                if (OUTBF16) outB[(size_t)row * N + col] = f2bf(val);
            }
        }
    }
}

// ---------------------------------------------------------------------------
// Scatter V into transposed layout vt[b,h,d,s] (bf16) from qkv[M, 3C].
// ---------------------------------------------------------------------------
__global__ __launch_bounds__(256) void scatter_vt(const __bf16* __restrict__ qkv,
                                                  __bf16* __restrict__ vt) {
    size_t idx = (size_t)blockIdx.x * 256 + threadIdx.x;  // (b,s,h,d), d fastest
    const int d = (int)(idx & 127);
    const int h = (int)((idx >> 7) & 15);
    const size_t bs = idx >> 11;                          // b*SEQL + s
    const int s = (int)(bs & (SEQL - 1));
    const int b = (int)(bs >> 11);
    vt[(((size_t)(b * NHEAD + h) * HDIM + d) * SEQL) + s] =
        qkv[bs * (3 * DMODEL) + 2 * DMODEL + (size_t)h * HDIM + d];
}

// ---------------------------------------------------------------------------
// Causal flash attention. One wave per 16-query tile. Q/K fragments read
// straight from the QKV buffer; V via vt. Scores scaled by 1/sqrt(HDIM).
// ---------------------------------------------------------------------------
__global__ __launch_bounds__(256, 1) void attention_wmma(
    const __bf16* __restrict__ qkv, const __bf16* __restrict__ vt,
    __bf16* __restrict__ y) {
    __shared__ __bf16 pbuf[8][16 * 32];  // per-wave P staging (C-layout -> A-layout)
    const int lane = threadIdx.x & 31;
    const int wave = threadIdx.x >> 5;
    const int l16 = lane & 15, lh = lane >> 4;

    const int tile = blockIdx.x * 8 + wave;   // 0..4095
    const int qt = tile & 127;                // query tile within sequence
    const int bh = tile >> 7;                 // b*NHEAD + h
    const int h = bh & 15, b = bh >> 4;

    // Q fragments (16 rows x 128 d = 4 K-chunks of 32)
    v16bf qf[4];
    {
        const __bf16* qp = qkv + (size_t)(b * SEQL + qt * 16 + l16) * (3 * DMODEL)
                               + h * HDIM + lh * 8;
        #pragma unroll
        for (int f = 0; f < 4; f++) qf[f] = load16(qp + f * 32, qp + f * 32 + 16);
    }

    v8f o[8] = {};
    float mrow[8], lrow[8];
    #pragma unroll
    for (int v = 0; v < 8; v++) { mrow[v] = -1e30f; lrow[v] = 0.f; }

    const float scale = 0.08838834764831845f;  // 1/sqrt(128)
    const int nk = ((qt * 16 + 16) + 31) >> 5; // 32-key chunks covering causal span
    __bf16* P = &pbuf[wave][0];

    for (int kt = 0; kt < nk; kt++) {
        // --- scores: two 16x16 tiles covering 32 keys ---
        v8f sc[2] = {};
        #pragma unroll
        for (int c = 0; c < 2; c++) {
            const __bf16* kp = qkv + (size_t)(b * SEQL + kt * 32 + c * 16 + l16) * (3 * DMODEL)
                                   + DMODEL + h * HDIM + lh * 8;
            #pragma unroll
            for (int f = 0; f < 4; f++) {
                v16bf kf = load16(kp + f * 32, kp + f * 32 + 16);
                sc[c] = __builtin_amdgcn_wmma_f32_16x16x32_bf16(
                    false, qf[f], false, kf, (short)0, sc[c], false, false);
            }
        }
        // --- online softmax (row stats per (vgpr, lane-half)) ---
        #pragma unroll
        for (int v = 0; v < 8; v++) {
            const int m = v + lh * 8;
            const int qs = qt * 16 + m;
            const int ks0 = kt * 32 + l16;
            float v0 = sc[0][v] * scale;
            float v1 = sc[1][v] * scale;
            if (ks0 > qs) v0 = -1e30f;
            if (ks0 + 16 > qs) v1 = -1e30f;
            float rmax = fmaxf(v0, v1);
            #pragma unroll
            for (int off = 8; off > 0; off >>= 1) rmax = fmaxf(rmax, __shfl_xor(rmax, off, 32));
            const float mnew = fmaxf(mrow[v], rmax);
            const float alpha = __expf(mrow[v] - mnew);
            const float p0 = __expf(v0 - mnew);
            const float p1 = __expf(v1 - mnew);
            float rsum = p0 + p1;
            #pragma unroll
            for (int off = 8; off > 0; off >>= 1) rsum += __shfl_xor(rsum, off, 32);
            lrow[v] = lrow[v] * alpha + rsum;
            mrow[v] = mnew;
            #pragma unroll
            for (int j = 0; j < 8; j++) o[j][v] *= alpha;
            P[m * 32 + l16]      = f2bf(p0);
            P[m * 32 + 16 + l16] = f2bf(p1);
        }
        // --- reload P as a 16x32 A-fragment ---
        const __bf16* pp = P + l16 * 32 + lh * 8;
        v16bf pf = load16(pp, pp + 16);
        // --- PV: 8 d-slabs of 16 ---
        #pragma unroll
        for (int j = 0; j < 8; j++) {
            const __bf16* vp = vt + ((size_t)bh * HDIM + j * 16 + l16) * SEQL
                                  + kt * 32 + lh * 8;
            v16bf vf = load16(vp, vp + 16);
            o[j] = __builtin_amdgcn_wmma_f32_16x16x32_bf16(
                false, pf, false, vf, (short)0, o[j], false, false);
        }
    }

    // Epilogue: y[b, s, h*HDIM + d] = o / l  (bf16)
    #pragma unroll
    for (int v = 0; v < 8; v++) {
        const int qs = qt * 16 + v + lh * 8;
        const float inv = 1.f / lrow[v];
        #pragma unroll
        for (int j = 0; j < 8; j++)
            y[(size_t)(b * SEQL + qs) * DMODEL + h * HDIM + j * 16 + l16] =
                f2bf(o[j][v] * inv);
    }
}

// ---------------------------------------------------------------------------
// Host orchestration
// ---------------------------------------------------------------------------
extern "C" void kernel_launch(void* const* d_in, const int* in_sizes, int n_in,
                              void* d_out, int out_size, void* d_ws, size_t ws_size,
                              hipStream_t stream) {
    const float* x      = (const float*)d_in[0];
    const float* ln1_w  = (const float*)d_in[1];
    const float* ln1_b  = (const float*)d_in[2];
    const float* qkv_w  = (const float*)d_in[3];
    const float* qkv_b  = (const float*)d_in[4];
    const float* out_w  = (const float*)d_in[5];
    const float* out_b  = (const float*)d_in[6];
    const float* ln2_w  = (const float*)d_in[7];
    const float* ln2_b  = (const float*)d_in[8];
    const float* mlp_w1 = (const float*)d_in[9];
    const float* mlp_b1 = (const float*)d_in[10];
    const float* mlp_w2 = (const float*)d_in[11];
    const float* mlp_b2 = (const float*)d_in[12];
    float* out = (float*)d_out;

    char* ws = (char*)d_ws;
    size_t off = 0;
    auto carve = [&](size_t bytes) -> char* {
        char* p = ws + off;
        off += (bytes + 255) & ~(size_t)255;
        return p;
    };
    __bf16* h1   = (__bf16*)carve((size_t)MROWS * DMODEL * 2);
    __bf16* qkvb = (__bf16*)carve((size_t)MROWS * 3 * DMODEL * 2);
    __bf16* vtb  = (__bf16*)carve((size_t)MROWS * DMODEL * 2);
    __bf16* yb   = (__bf16*)carve((size_t)MROWS * DMODEL * 2);
    float*  x2   = (float*) carve((size_t)MROWS * DMODEL * 4);
    __bf16* h2   = (__bf16*)carve((size_t)MROWS * DMODEL * 2);
    __bf16* mb   = (__bf16*)carve((size_t)MROWS * DFF * 2);
    __bf16* wq   = (__bf16*)carve((size_t)3 * DMODEL * DMODEL * 2);
    __bf16* wo   = (__bf16*)carve((size_t)DMODEL * DMODEL * 2);
    __bf16* w1   = (__bf16*)carve((size_t)DFF * DMODEL * 2);
    __bf16* w2   = (__bf16*)carve((size_t)DMODEL * DFF * 2);

    // 1) weights -> bf16
    {
        size_t n;
        n = (size_t)3 * DMODEL * DMODEL;
        cvt_f32_bf16<<<(int)((n + 255) / 256), 256, 0, stream>>>(qkv_w, wq, n);
        n = (size_t)DMODEL * DMODEL;
        cvt_f32_bf16<<<(int)((n + 255) / 256), 256, 0, stream>>>(out_w, wo, n);
        n = (size_t)DFF * DMODEL;
        cvt_f32_bf16<<<(int)((n + 255) / 256), 256, 0, stream>>>(mlp_w1, w1, n);
        n = (size_t)DMODEL * DFF;
        cvt_f32_bf16<<<(int)((n + 255) / 256), 256, 0, stream>>>(mlp_w2, w2, n);
    }

    // 2) LN1 -> h1 (bf16)
    layernorm_bf16<<<MROWS, 256, 0, stream>>>(x, ln1_w, ln1_b, h1, DMODEL);

    // 3) QKV GEMM: [4096,2048] x [6144,2048]^T -> qkvb (bf16)
    gemm_wmma_bf16<0, false, false, true><<<dim3(3 * DMODEL / 128, MROWS / 128), 256, 0, stream>>>(
        h1, wq, qkv_b, nullptr, nullptr, qkvb, MROWS, 3 * DMODEL, DMODEL);

    // 4) transpose V -> vt[b,h,d,s]
    scatter_vt<<<(int)(((size_t)MROWS * DMODEL + 255) / 256), 256, 0, stream>>>(qkvb, vtb);

    // 5) flash attention -> yb (bf16)
    attention_wmma<<<(MROWS / 16) / 8 * (NHEAD * BATCH) / (NHEAD * BATCH) * ((BATCH * NHEAD * (SEQL / 16)) / 8),
                    256, 0, stream>>>(qkvb, vtb, yb);

    // 6) out-proj + residual: x2 = x + yb @ out_w^T + out_b  (fp32)
    gemm_wmma_bf16<0, true, true, false><<<dim3(DMODEL / 128, MROWS / 128), 256, 0, stream>>>(
        yb, wo, out_b, x, x2, nullptr, MROWS, DMODEL, DMODEL);

    // 7) LN2 -> h2 (bf16)
    layernorm_bf16<<<MROWS, 256, 0, stream>>>(x2, ln2_w, ln2_b, h2, DMODEL);

    // 8) MLP up + exact GELU -> mb (bf16)
    gemm_wmma_bf16<1, false, false, true><<<dim3(DFF / 128, MROWS / 128), 256, 0, stream>>>(
        h2, w1, mlp_b1, nullptr, nullptr, mb, MROWS, DFF, DMODEL);

    // 9) MLP down + residual -> out (fp32)
    gemm_wmma_bf16<0, true, true, false><<<dim3(DMODEL / 128, MROWS / 128), 256, 0, stream>>>(
        mb, w2, mlp_b2, x2, out, nullptr, MROWS, DMODEL, DFF);

    (void)in_sizes; (void)n_in; (void)out_size; (void)ws_size;
}
